// ResonanceMemoryAttentionV15_54571854463351
// MI455X (gfx1250) — matrix operand order repaired and checked by reading, not compiled, not run on hardware
//
#include <hip/hip_runtime.h>
#include <math.h>

typedef __attribute__((ext_vector_type(16))) _Float16 v16h;
typedef __attribute__((ext_vector_type(8)))  _Float16 v8h;
typedef __attribute__((ext_vector_type(8)))  float    v8f;
typedef __attribute__((ext_vector_type(4)))  int      v4i;

#define Bn   4
#define Ln   512
#define Dn   768
#define Hn   12
#define DHn  64
#define Rn   128
#define ROWS (Bn*Ln)     /* 2048 */
#define NQKV (3*Dn)      /* 2304 */

#if defined(__HIP_DEVICE_COMPILE__) && \
    __has_builtin(__builtin_amdgcn_global_load_async_to_lds_b128) && \
    __has_builtin(__builtin_amdgcn_s_wait_asynccnt)
#define RMA_ASYNC_LDS 1
#define GLOBAL_V4I(p) ((__attribute__((address_space(1))) v4i*)(p))
#define LDS_V4I(p)    ((__attribute__((address_space(3))) v4i*)(p))
#else
#define RMA_ASYNC_LDS 0
#endif

__device__ __forceinline__ float sigmoidf_(float x){ return 1.f/(1.f+__expf(-x)); }

// ---------------- LayerNorm over D=768, emit f16 A-matrix for QKV GEMM ----
__global__ void rma_ln_kernel(const float* __restrict__ x, const float* __restrict__ g,
                              const float* __restrict__ b, _Float16* __restrict__ xn){
  int row = blockIdx.x;
  const float* xr = x + (size_t)row*Dn;
  __shared__ float red[256];
  int t = threadIdx.x;
  float s = 0.f;
  for (int d = t; d < Dn; d += 256) s += xr[d];
  red[t] = s; __syncthreads();
  for (int k = 128; k > 0; k >>= 1){ if (t < k) red[t] += red[t+k]; __syncthreads(); }
  float mu = red[0] * (1.f/Dn);
  __syncthreads();
  float s2 = 0.f;
  for (int d = t; d < Dn; d += 256){ float v = xr[d]-mu; s2 += v*v; }
  red[t] = s2; __syncthreads();
  for (int k = 128; k > 0; k >>= 1){ if (t < k) red[t] += red[t+k]; __syncthreads(); }
  float rsig = rsqrtf(red[0]*(1.f/Dn) + 1e-5f);
  for (int d = t; d < Dn; d += 256)
    xn[(size_t)row*Dn + d] = (_Float16)((xr[d]-mu)*rsig*g[d] + b[d]);
}

// ---------------- Bottleneck: silu(x@W1)@W2 -> gate/decay ------------------
__global__ void rma_bottleneck_kernel(const float* __restrict__ x, const float* __restrict__ W1,
                                      const float* __restrict__ W2, const float* __restrict__ temp,
                                      float* __restrict__ gate_out, float* __restrict__ decayb,
                                      float* __restrict__ gateb){
  __shared__ float xr[Dn];
  __shared__ float h1[Rn];
  __shared__ float p[Hn*5];
  int row = blockIdx.x;
  int t = threadIdx.x;                       // 128 threads
  for (int d = t; d < Dn; d += 128) xr[d] = x[(size_t)row*Dn + d];
  __syncthreads();
  float acc = 0.f;
  for (int d = 0; d < Dn; d++) acc += xr[d] * W1[d*Rn + t];
  h1[t] = acc * sigmoidf_(acc);              // silu
  __syncthreads();
  if (t < Hn*5){
    float s = 0.f;
    for (int r = 0; r < Rn; r++) s += h1[r] * W2[r*(Hn*5) + t];
    p[t] = s;
  }
  __syncthreads();
  if (t < Hn){
    float sa = sigmoidf_(p[t*5+0]);
    float sp = tanhf(p[t*5+1]) * 3.14159265358979323846f;
    float ca = sigmoidf_(p[t*5+2]);
    float cp = tanhf(p[t*5+3]) * 3.14159265358979323846f;
    float dec = 0.5f + 0.49f * sigmoidf_(p[t*5+4]);
    float tm  = fminf(fmaxf(temp[0], 0.1f), 2.0f);
    float gt  = sigmoidf_(tanhf(sa*ca*__cosf(sp-cp)) * tm);
    gate_out[row*Hn + t] = gt;
    gateb[row*Hn + t]    = gt;
    decayb[row*Hn + t]   = dec;
  }
}

// ---------------- WMMA GEMM: C(MxN) = A_f16(MxK) @ W_f32(KxN) + bias -------
// Block tile 128x64, 8 waves; each wave owns a 16x64 strip (4 WMMAs / K-step).
// B is stored transposed in LDS so A and B fragments are contiguous b128 loads.
// mode 0: qkv epilogue (elu+1 on first 2*Dn columns), mode 1: plain
__global__ void rma_gemm_kernel(const _Float16* __restrict__ A, const float* __restrict__ W,
                                const float* __restrict__ bias, float* __restrict__ out,
                                int N, int K, int mode){
  __shared__ __align__(16) _Float16 As[128][40];   // 80B pitch (16B multiple)
  __shared__ __align__(16) _Float16 BsT[64][40];   // [col][k], 80B pitch
  int t    = threadIdx.x;
  int lane = t & 31;
  int wave = t >> 5;              // 0..7 -> 16-row strip
  int m0   = blockIdx.y * 128;
  int n0   = blockIdx.x * 64;

  v8f acc[4] = {v8f{}, v8f{}, v8f{}, v8f{}};

  int ar = t >> 1;                // 0..127
  int ac = (t & 1) * 16;          // 0 or 16
  int br = t >> 3;                // 0..31 (k row)
  int bc = (t & 7) * 8;           // 0..56 (col chunk)

  for (int kc = 0; kc < K; kc += 32){
    // ---- A tile 128x32 (f16, 2x16B per thread) ----
    const _Float16* asrc = A + (size_t)(m0 + ar)*K + kc + ac;
#if RMA_ASYNC_LDS
    __builtin_amdgcn_global_load_async_to_lds_b128(
        GLOBAL_V4I(asrc),     LDS_V4I(&As[ar][ac]),     0, 0);
    __builtin_amdgcn_global_load_async_to_lds_b128(
        GLOBAL_V4I(asrc + 8), LDS_V4I(&As[ar][ac + 8]), 0, 0);
#else
    *(v8h*)&As[ar][ac]     = *(const v8h*)asrc;
    *(v8h*)&As[ar][ac + 8] = *(const v8h*)(asrc + 8);
#endif
    // ---- B tile 32x64 (f32 -> f16, transposed into BsT[col][k]) ----
    {
      const float4* bsrc = (const float4*)(W + (size_t)(kc + br)*N + n0 + bc);
      float4 w0 = bsrc[0];
      float4 w1 = bsrc[1];
      BsT[bc+0][br] = (_Float16)w0.x;  BsT[bc+1][br] = (_Float16)w0.y;
      BsT[bc+2][br] = (_Float16)w0.z;  BsT[bc+3][br] = (_Float16)w0.w;
      BsT[bc+4][br] = (_Float16)w1.x;  BsT[bc+5][br] = (_Float16)w1.y;
      BsT[bc+6][br] = (_Float16)w1.z;  BsT[bc+7][br] = (_Float16)w1.w;
    }
#if RMA_ASYNC_LDS
    __builtin_amdgcn_s_wait_asynccnt(0);
#endif
    __syncthreads();

    // A fragment (16x32): lane<16 -> K {0..7,16..23}; lane>=16 -> K {8..15,24..31}
    int arow = wave*16 + (lane & 15);
    int kb   = (lane < 16) ? 0 : 8;
    v8h a_lo = *(const v8h*)&As[arow][kb];
    v8h a_hi = *(const v8h*)&As[arow][16 + kb];
    v16h af  = __builtin_shufflevector(a_lo, a_hi, 0,1,2,3,4,5,6,7,8,9,10,11,12,13,14,15);

    // B fragments (32x16): lane<16 -> K {0..7,16..23} of col lane; lane>=16 -> K {8..15,24..31}
    int kb2 = (lane < 16) ? 0 : 8;
    int bcl = lane & 15;
    #pragma unroll
    for (int j = 0; j < 4; j++){
      v8h b_lo = *(const v8h*)&BsT[j*16 + bcl][kb2];
      v8h b_hi = *(const v8h*)&BsT[j*16 + bcl][16 + kb2];
      v16h bf  = __builtin_shufflevector(b_lo, b_hi, 0,1,2,3,4,5,6,7,8,9,10,11,12,13,14,15);
      acc[j] = __builtin_amdgcn_wmma_f32_16x16x32_f16(false, af, false, bf, (short)0, acc[j], false, false);
    }
    __syncthreads();
  }

  // C layout: VGPR r -> M = r (lane<16) / 8+r (lane>=16), N = lane&15
  int rbase = (lane < 16) ? 0 : 8;
  #pragma unroll
  for (int j = 0; j < 4; j++){
    int gn = n0 + j*16 + (lane & 15);
    float bi = bias[gn];
    bool transform = (mode == 0) && (gn < 2*Dn);
    #pragma unroll
    for (int r = 0; r < 8; r++){
      int gm = m0 + wave*16 + rbase + r;
      float c = acc[j][r] + bi;
      float ce = __expf(fminf(c, 0.f));            // elu(c)+1 for c<=0
      c = transform ? ((c > 0.f) ? (c + 1.f) : ce) : c;
      out[(size_t)gm*N + gn] = c;
    }
  }
}

// ---------------- Decayed linear-attention scan + fused head layernorm -----
// One block per (b,h); 64x64 state in registers (16 floats/thread).
__global__ void rma_scan_kernel(const float* __restrict__ qkv, const float* __restrict__ decayb,
                                const float* __restrict__ gateb, const float* __restrict__ mg,
                                const float* __restrict__ mb, _Float16* __restrict__ attnA){
  int bh = blockIdx.x;
  int b  = bh / Hn, h = bh % Hn;
  int t  = threadIdx.x;          // 256
  int e  = t & 63;               // state column
  int rg = t >> 6;               // row group (16 rows each)
  __shared__ float kf[64], vf[64], qf[64];
  __shared__ float red[4][64];
  __shared__ float dpA[64], outs[64], outs2[64];
  __shared__ float scal[2];
  float S[16];
  #pragma unroll
  for (int i = 0; i < 16; i++) S[i] = 0.f;
  float z = 0.f;
  float cld = 0.f;
  float gg = (t < 64) ? mg[e] : 0.f;
  float gb = (t < 64) ? mb[e] : 0.f;

  for (int l = 0; l < Ln; l++){
    int row = b*Ln + l;
    if (t < 64){
      qf[e] = qkv[(size_t)row*NQKV +          h*DHn + e];
      kf[e] = qkv[(size_t)row*NQKV +   Dn   + h*DHn + e];
      vf[e] = qkv[(size_t)row*NQKV + 2*Dn   + h*DHn + e];
    }
    if (t == 0){
      scal[0] = decayb[row*Hn + h];
      scal[1] = gateb[row*Hn + h];
    }
    __syncthreads();
    float dec = scal[0];
    float gt  = scal[1];
    // reproduce reference's clamp: cld = max(cumsum(log(decay+1e-8)), -85)
    float ld   = __logf(dec + 1e-8f);
    float cldn = cld + ld;
    float eff  = __expf(fmaxf(cldn, -85.f) - fmaxf(cld, -85.f));
    cld = cldn;
    float ema  = 1.f - dec;
    float coef = ema * (1.f + gt);
    float ve   = vf[e];
    float part = 0.f;
    #pragma unroll
    for (int i = 0; i < 16; i++){
      int d = rg*16 + i;
      S[i] = S[i]*eff + coef * kf[d] * ve;
      part += qf[d] * S[i];
    }
    red[rg][e] = part;
    if (t < 64){
      z = z*eff + ema * kf[t];
      dpA[t] = qf[t] * z;
    }
    __syncthreads();
    float num = (t < 64) ? (red[0][e] + red[1][e] + red[2][e] + red[3][e]) : 0.f;
    for (int k2 = 32; k2 > 0; k2 >>= 1){ if (t < k2) dpA[t] += dpA[t+k2]; __syncthreads(); }
    float den = dpA[0] + 1e-6f;
    float o = num / den;
    if (t < 64){ outs[e] = o; outs2[e] = o*o; }
    __syncthreads();
    for (int k2 = 32; k2 > 0; k2 >>= 1){
      if (t < k2){ outs[t] += outs[t+k2]; outs2[t] += outs2[t+k2]; }
      __syncthreads();
    }
    float mu  = outs[0]  * (1.f/64.f);
    float var = outs2[0] * (1.f/64.f) - mu*mu;
    float rsg = rsqrtf(var + 1e-5f);
    if (t < 64)
      attnA[(size_t)row*Dn + h*DHn + e] = (_Float16)((o - mu)*rsg*gg + gb);
    __syncthreads();
  }
}

extern "C" void kernel_launch(void* const* d_in, const int* in_sizes, int n_in,
                              void* d_out, int out_size, void* d_ws, size_t ws_size,
                              hipStream_t stream){
  (void)in_sizes; (void)n_in; (void)out_size; (void)ws_size;
  const float* x      = (const float*)d_in[0];
  const float* W_qkv  = (const float*)d_in[1];
  const float* b_qkv  = (const float*)d_in[2];
  const float* W_b1   = (const float*)d_in[3];
  const float* W_b2   = (const float*)d_in[4];
  const float* temp   = (const float*)d_in[5];
  const float* W_proj = (const float*)d_in[6];
  const float* b_proj = (const float*)d_in[7];
  const float* ln_g   = (const float*)d_in[8];
  const float* ln_b   = (const float*)d_in[9];
  const float* mem_g  = (const float*)d_in[10];
  const float* mem_b  = (const float*)d_in[11];

  float* out      = (float*)d_out;                     // (B,L,D) projection output
  float* gate_out = out + (size_t)ROWS*Dn;             // (B,L,H) gate output

  char* ws = (char*)d_ws;
  float*    qkvT   = (float*)ws;    ws += (size_t)ROWS*NQKV*sizeof(float);   // 18.9 MB
  float*    decayb = (float*)ws;    ws += (size_t)ROWS*Hn*sizeof(float);
  float*    gateb  = (float*)ws;    ws += (size_t)ROWS*Hn*sizeof(float);
  _Float16* xn     = (_Float16*)ws; ws += (size_t)ROWS*Dn*sizeof(_Float16);  // 3 MB
  _Float16* attnA  = (_Float16*)ws; ws += (size_t)ROWS*Dn*sizeof(_Float16);  // 3 MB

  rma_ln_kernel<<<ROWS, 256, 0, stream>>>(x, ln_g, ln_b, xn);
  rma_bottleneck_kernel<<<ROWS, 128, 0, stream>>>(x, W_b1, W_b2, temp, gate_out, decayb, gateb);
  rma_gemm_kernel<<<dim3(NQKV/64, ROWS/128), 256, 0, stream>>>(xn, W_qkv, b_qkv, qkvT, NQKV, Dn, 0);
  rma_scan_kernel<<<Bn*Hn, 256, 0, stream>>>(qkvT, decayb, gateb, mem_g, mem_b, attnA);
  rma_gemm_kernel<<<dim3(Dn/64, ROWS/128), 256, 0, stream>>>(attnA, W_proj, b_proj, out, Dn, Dn, 1);
}
